// SPoT_19464791785499
// MI455X (gfx1250) — compile-verified, matrix operand order checked by reading, not run on hardware
//
#include <hip/hip_runtime.h>
#include <hip/hip_bf16.h>
#include <math.h>
#include <stdint.h>

// ---------------- model constants ----------------
#define BB    32
#define NTOK  257            // cls + 256
#define RROWS (BB * NTOK)    // 8224 = 514*16
#define MPAD  8320           // 65*128: padded rows for staged GEMM A tiles
#define EE    384
#define HID3  1536
#define QKVN  1152
#define IND   2304           // (3+1)*24*24
#define NF    256
#define HEADS 6
#define DH    64
#define NPAD  288            // 257 padded to 9*32 for WMMA K-loops
#define RT_PER_BH 17         // ceil(257/16)

#if __has_builtin(__builtin_amdgcn_tensor_load_to_lds)
#define HAVE_TDM 1
#else
#define HAVE_TDM 0
#endif

typedef __attribute__((ext_vector_type(16))) __bf16 v16bf;
typedef __attribute__((ext_vector_type(8)))  float  v8f;
typedef __attribute__((ext_vector_type(4)))  unsigned int u32x4;
typedef __attribute__((ext_vector_type(8)))  int i32x8;
typedef __attribute__((ext_vector_type(4)))  int i32x4;

union FragBF { v16bf v; unsigned u[8]; };

__device__ __forceinline__ unsigned short f32_to_bf16(float f) {
    unsigned u = __float_as_uint(f);
    u += 0x7FFFu + ((u >> 16) & 1u);   // round to nearest even
    return (unsigned short)(u >> 16);
}

// ================= TDM: 2-D tensor tile -> LDS (CDNA5 Tensor Data Mover) =====
// D# per ISA 8.3/8.4: group0 = {count=1, lds_addr, global_addr[56:0], type=2};
// group1 = {mask=0, data_size=1(2B), tensor_dim0/1, tile_dim0/1, dim0_stride}.
__device__ __forceinline__ void tdm_load_2d(const void* gaddr, unsigned lds_off,
                                            unsigned tensor_d0, unsigned tensor_d1,
                                            unsigned tile_d0, unsigned tile_d1,
                                            unsigned long long stride0_elems) {
#if HAVE_TDM
    const unsigned long long ga = (unsigned long long)(uintptr_t)gaddr;
    u32x4 g0;
    g0.x = 1u;                                        // count=1, user desc
    g0.y = lds_off;                                   // lds_addr (bytes)
    g0.z = (unsigned)(ga & 0xFFFFFFFFu);              // global_addr[31:0]
    g0.w = (unsigned)((ga >> 32) & 0x01FFFFFFu) | (2u << 30);  // [56:32] | type=2
    i32x8 g1;
    g1[0] = (int)(1u << 16);                          // data_size=1 -> 2 bytes
    g1[1] = (int)((tensor_d0 & 0xFFFFu) << 16);       // tensor_dim0[15:0]
    g1[2] = (int)((tensor_d0 >> 16) | ((tensor_d1 & 0xFFFFu) << 16));
    g1[3] = (int)((tensor_d1 >> 16) | ((tile_d0 & 0xFFFFu) << 16));
    g1[4] = (int)(tile_d1 & 0xFFFFu);                 // tile_dim1, tile_dim2=0
    g1[5] = (int)(stride0_elems & 0xFFFFFFFFull);     // dim0_stride[31:0]
    g1[6] = (int)((stride0_elems >> 32) & 0xFFFFull); // dim0_stride[47:32]
    g1[7] = 0;
    i32x4 z4 = {};
#if __clang_major__ >= 23
    i32x8 z8 = {};
    __builtin_amdgcn_tensor_load_to_lds(g0, g1, z4, z4, z8, 0);
#else
    __builtin_amdgcn_tensor_load_to_lds(g0, g1, z4, z4, 0);
#endif
#endif
}

// ---- A fragment from LDS tile [128][32] (16-bit A 16x32 VGPR layout) ----
__device__ __forceinline__ v16bf load_frag_A_lds(const unsigned short* sA, int wave) {
    const int lane = threadIdx.x & 31;
    const int g = lane >> 4;
    const unsigned short* row = sA + (wave * 16 + (lane & 15)) * 32;
    FragBF f;
#pragma unroll
    for (int v = 0; v < 8; ++v) {
        const int h = v >> 2, j = v & 3;
        f.u[v] = *(const unsigned*)(row + 16 * h + 8 * g + 2 * j); // -> ds_load_b128 x2
    }
    return f.v;
}

// ---- B fragment from pre-packed weights: one 32B contiguous load per lane ----
__device__ __forceinline__ v16bf load_frag_Bp(const unsigned short* P, int kdiv32,
                                              int kc, int nt) {
    const int lane = threadIdx.x & 31;
    const size_t off = ((size_t)(nt * kdiv32 + kc) * 32 + lane) * 16;
    return *(const v16bf*)(P + off);
}

// ---------------- weight pack: f32 [K x N] -> fragment-ordered bf16 ----------
__global__ void k_pack_B(const float* __restrict__ W, unsigned short* __restrict__ P,
                         int K, int N) {
    const int i = blockIdx.x * blockDim.x + threadIdx.x;
    if (i >= K * N) return;
    const int e = i & 15;
    const int lane = (i >> 4) & 31;
    const int fb = i >> 9;
    const int kdiv32 = K >> 5;
    const int kc = fb % kdiv32;
    const int nt = fb / kdiv32;
    const int g = lane >> 4;
    const int n = nt * 16 + (lane & 15);
    const int v = e >> 1, pr = e & 1;
    const int h = v >> 2, j = v & 3;
    const int k = kc * 32 + 16 * h + 8 * g + 2 * j + pr;
    P[i] = f32_to_bf16(W[(size_t)k * N + n]);
}

// ---------------- cls row init ----------------
__global__ void k_init_cls(const float* __restrict__ cls, float* __restrict__ x) {
    const int i = blockIdx.x * blockDim.x + threadIdx.x;
    if (i < BB * EE) {
        const int b = i / EE, e = i % EE;
        x[(size_t)b * NTOK * EE + e] = cls[e];
    }
}

// ---------------- token extraction (gather + bilinear + gauss) ----------------
__global__ void k_extract_tokens(const float* __restrict__ img,
                                 const float* __restrict__ mu,
                                 unsigned short* __restrict__ feats) {
    const int tok = blockIdx.x;            // 0..8191
    const int b = tok >> 8;
    const float mux = mu[(size_t)tok * 2 + 0];
    const float muy = mu[(size_t)tok * 2 + 1];
    const float cx = (mux + 1.f) * 0.5f * 223.f;
    const float cy = (muy + 1.f) * 0.5f * 223.f;
    const float* im = img + (size_t)b * 3 * 224 * 224;
    unsigned short* row = feats + (size_t)tok * IND;
    for (int p = threadIdx.x; p < 576; p += blockDim.x) {
        const int i = p / 24, j = p % 24;
        const float gj = -1.f + (2.f / 23.f) * j;
        const float gi = -1.f + (2.f / 23.f) * i;
        float px = cx + gj * 11.5f; px = fminf(fmaxf(px, 0.f), 223.f);
        float py = cy + gi * 11.5f; py = fminf(fmaxf(py, 0.f), 223.f);
        float fx0 = floorf(px); if (fx0 > 222.f) fx0 = 222.f;
        float fy0 = floorf(py); if (fy0 > 222.f) fy0 = 222.f;
        const int x0 = (int)fx0, y0 = (int)fy0;
        const float wx = px - fx0, wy = py - fy0;
        const float gauss = __expf(-1.125f * (gj * gj + gi * gi)); // 0.5*1.5^2
        const int i00 = y0 * 224 + x0;
#pragma unroll
        for (int c = 0; c < 3; ++c) {
            const float* ic = im + c * 224 * 224;
            const float v00 = ic[i00], v01 = ic[i00 + 1];
            const float v10 = ic[i00 + 224], v11 = ic[i00 + 225];
            const float bil = v00 * (1.f - wx) * (1.f - wy) + v01 * wx * (1.f - wy)
                            + v10 * (1.f - wx) * wy + v11 * wx * wy;
            row[c * 576 + p] = f32_to_bf16(bil * gauss);
        }
        const float pos = 0.5f * (px * (1.f / 223.f) + py * (1.f / 223.f));
        row[3 * 576 + p] = f32_to_bf16(pos);
    }
}

// ------- bf16 WMMA GEMM: double-buffered TDM-staged A, packed B, epilogues ----
// Block tile 128(M) x 128(N); 8 waves each own 16 rows x 8 column fragments.
// EPI 0: embed    -> outF[(m + m/256 + 1)*ldo + n] = acc + bias
// EPI 2: residual -> outF[m*ldo + n] += (acc + bias) * ls[n]
// EPI 3: gelu     -> outH[m*ldo + n] = bf16(gelu(acc + bias))
// EPI 4: qkv      -> outH row-major for Q,K; V transposed into outH2 [B,H,64,288]
template <int EPI>
__global__ void k_gemm_bf16(const unsigned short* __restrict__ A,
                            const unsigned short* __restrict__ Bp,
                            const float* __restrict__ bias,
                            const float* __restrict__ ls,
                            float* __restrict__ outF,
                            unsigned short* __restrict__ outH,
                            unsigned short* __restrict__ outH2,
                            int M, int N, int K, int lda, int ldo, int mpad) {
    __shared__ unsigned short sA[2][128 * 32];
    const int wave = threadIdx.x >> 5;                    // 8 waves/block
    const int blk_m = blockIdx.y * 128;
    const int tile_m = blk_m + wave * 16;
    const int tile_n = blockIdx.x * 128;
    const int kdiv32 = K >> 5;
    const unsigned rowsAvail = (unsigned)(mpad - blk_m);
    v8f acc[8] = {};

#if HAVE_TDM
    if (wave == 0) {                     // prologue: stage chunk 0 into buffer 0
        tdm_load_2d(A + (size_t)blk_m * lda,
                    (unsigned)(uintptr_t)&sA[0][0],
                    (unsigned)lda, rowsAvail, 32u, 128u,
                    (unsigned long long)lda);
    }
#endif
    for (int kc = 0; kc < kdiv32; ++kc) {
        const int cur = kc & 1;
#if HAVE_TDM
        if (wave == 0) {
            if (kc + 1 < kdiv32) {       // issue next chunk into the other buffer
                tdm_load_2d(A + (size_t)blk_m * lda + (kc + 1) * 32,
                            (unsigned)(uintptr_t)&sA[cur ^ 1][0],
                            (unsigned)lda, rowsAvail, 32u, 128u,
                            (unsigned long long)lda);
#if __has_builtin(__builtin_amdgcn_s_wait_tensorcnt)
                __builtin_amdgcn_s_wait_tensorcnt(1);   // in-order: current landed
#endif
            } else {
#if __has_builtin(__builtin_amdgcn_s_wait_tensorcnt)
                __builtin_amdgcn_s_wait_tensorcnt(0);
#endif
            }
        }
#else
        for (int id = threadIdx.x; id < 2048; id += blockDim.x) {   // 8KB as u32
            const int row = id >> 3, c = id & 7;
            *(unsigned*)(&sA[cur][0] + row * 32 + c * 2) =
                *(const unsigned*)(A + (size_t)(blk_m + row) * lda + kc * 32 + c * 2);
        }
#endif
        __syncthreads();                 // staged tile visible to all waves
        __builtin_prefetch(Bp + ((size_t)((tile_n >> 4) * kdiv32 + kc + 1) * 32
                                 + (threadIdx.x & 31)) * 16, 0, 1);
        v16bf a = load_frag_A_lds(&sA[cur][0], wave);
#pragma unroll
        for (int t = 0; t < 8; ++t) {
            v16bf bfr = load_frag_Bp(Bp, kdiv32, kc, (tile_n >> 4) + t);
            acc[t] = __builtin_amdgcn_wmma_f32_16x16x32_bf16(
                false, a, false, bfr, (short)0, acc[t], false, false);
        }
        __syncthreads();                 // all reads done before buffer reuse
    }
    const int lane = threadIdx.x & 31;
    const int g = lane >> 4;
#pragma unroll
    for (int t = 0; t < 8; ++t) {
        const int n = tile_n + 16 * t + (lane & 15);
        const float bv = bias ? bias[n] : 0.f;
#pragma unroll
        for (int r = 0; r < 8; ++r) {
            const int m = tile_m + 8 * g + r;
            if (m >= M) continue;
            const float v = acc[t][r] + bv;
            if (EPI == 0) {
                const int row = m + m / NF + 1;           // b*257 + 1 + t
                outF[(size_t)row * ldo + n] = v;
            } else if (EPI == 2) {
                outF[(size_t)m * ldo + n] += v * ls[n];
            } else if (EPI == 3) {
                const float gl = 0.5f * v * (1.f + erff(v * 0.70710678f));
                outH[(size_t)m * ldo + n] = f32_to_bf16(gl);
            } else {                                      // EPI == 4 (qkv)
                if (n < 2 * EE) {
                    outH[(size_t)m * ldo + n] = f32_to_bf16(v);
                } else {
                    const int dg = n - 2 * EE;
                    const int hh = dg >> 6, d = dg & 63;
                    const int bq = m / NTOK, tk = m % NTOK;
                    outH2[((size_t)(bq * HEADS + hh) * DH + d) * NPAD + tk] =
                        f32_to_bf16(v);
                }
            }
        }
    }
}

// ---------------- layernorm (one row / block, 128 threads x 3 elems) ----------
template <bool OUT_BF16>
__global__ void k_layernorm(const float* __restrict__ x,
                            const float* __restrict__ gw,
                            const float* __restrict__ bw,
                            unsigned short* __restrict__ outH,
                            float* __restrict__ outF) {
    const int row = blockIdx.x;
    const int tid = threadIdx.x;
    const float* xr = x + (size_t)row * EE;
    __shared__ float red[128];
    const float v0 = xr[tid], v1 = xr[tid + 128], v2 = xr[tid + 256];
    red[tid] = v0 + v1 + v2;
    __syncthreads();
    for (int off = 64; off > 0; off >>= 1) {
        if (tid < off) red[tid] += red[tid + off];
        __syncthreads();
    }
    const float mean = red[0] * (1.f / EE);
    __syncthreads();
    const float d0 = v0 - mean, d1 = v1 - mean, d2 = v2 - mean;
    red[tid] = d0 * d0 + d1 * d1 + d2 * d2;
    __syncthreads();
    for (int off = 64; off > 0; off >>= 1) {
        if (tid < off) red[tid] += red[tid + off];
        __syncthreads();
    }
    const float rstd = rsqrtf(red[0] * (1.f / EE) + 1e-6f);
#pragma unroll
    for (int s = 0; s < 3; ++s) {
        const int e = tid + 128 * s;
        const float d = (s == 0 ? d0 : (s == 1 ? d1 : d2));
        const float o = d * rstd * gw[e] + bw[e];
        if (OUT_BF16) outH[(size_t)row * EE + e] = f32_to_bf16(o);
        else          outF[(size_t)row * EE + e] = o;
    }
}

// ---------------- attention: one wave per (b, h, 16-row q tile) ---------------
__global__ void k_attention(const unsigned short* __restrict__ qkv,  // [R,1152]
                            const unsigned short* __restrict__ vt,   // [B,H,64,288]
                            unsigned short* __restrict__ o) {        // [R,384]
    int id = blockIdx.x;
    const int rt = id % RT_PER_BH; id /= RT_PER_BH;
    const int h = id % HEADS;
    const int b = id / HEADS;
    __shared__ float          sS[16 * NPAD];
    __shared__ unsigned short sP[16 * NPAD];
    const int lane = threadIdx.x & 31;
    const int g = lane >> 4;
    const size_t base = (size_t)b * NTOK * QKVN + h * DH;

    // ---- S = Q * K^T (K=64 -> 2 chunks) ----
    for (int ct = 0; ct < NPAD / 16; ++ct) {
        v8f acc = {};
#pragma unroll
        for (int kc = 0; kc < 2; ++kc) {
            int mrow = rt * 16 + (lane & 15); if (mrow > NTOK - 1) mrow = NTOK - 1;
            const unsigned short* qr = qkv + base + (size_t)mrow * QKVN + kc * 32;
            FragBF fa;
#pragma unroll
            for (int v = 0; v < 8; ++v) {
                const int hh = v >> 2, j = v & 3;
                fa.u[v] = *(const unsigned*)(qr + 16 * hh + 8 * g + 2 * j);
            }
            int ncol = ct * 16 + (lane & 15); if (ncol > NTOK - 1) ncol = NTOK - 1;
            const unsigned short* kr = qkv + base + EE + (size_t)ncol * QKVN + kc * 32;
            FragBF fb;
#pragma unroll
            for (int v = 0; v < 8; ++v) {
                const int hh = v >> 2, j = v & 3;
                fb.u[v] = *(const unsigned*)(kr + 16 * hh + 8 * g + 2 * j);
            }
            acc = __builtin_amdgcn_wmma_f32_16x16x32_bf16(
                false, fa.v, false, fb.v, (short)0, acc, false, false);
        }
#pragma unroll
        for (int r = 0; r < 8; ++r)
            sS[(8 * g + r) * NPAD + ct * 16 + (lane & 15)] = acc[r];
    }
    __syncthreads();

    // ---- softmax over 257 valid keys, scale = D^-0.5 = 0.125 ----
    if (lane < 16) {
        float* srow = sS + lane * NPAD;
        unsigned short* prow = sP + lane * NPAD;
        float mx = -1e30f;
        for (int j = 0; j < NTOK; ++j) {
            const float v = srow[j] * 0.125f;
            srow[j] = v;
            mx = fmaxf(mx, v);
        }
        float sum = 0.f;
        for (int j = 0; j < NTOK; ++j) {
            const float e = __expf(srow[j] - mx);
            srow[j] = e;
            sum += e;
        }
        const float inv = 1.f / sum;
        for (int j = 0; j < NTOK; ++j) prow[j] = f32_to_bf16(srow[j] * inv);
        for (int j = NTOK; j < NPAD; ++j) prow[j] = 0;
    }
    __syncthreads();

    // ---- O = P * V   (V pre-transposed: contiguous K per lane) ----
    const unsigned short* vbase = vt + (size_t)(b * HEADS + h) * DH * NPAD;
    v8f oacc[4] = {};
    for (int kc = 0; kc < NPAD / 32; ++kc) {
        FragBF fa;
        const int mrow = lane & 15;
#pragma unroll
        for (int v = 0; v < 8; ++v) {
            const int hh = v >> 2, j = v & 3;
            fa.u[v] = *(const unsigned*)(sP + mrow * NPAD + kc * 32
                                         + 16 * hh + 8 * g + 2 * j);
        }
#pragma unroll
        for (int t = 0; t < 4; ++t) {
            const int n = t * 16 + (lane & 15);
            const unsigned short* vr = vbase + (size_t)n * NPAD + kc * 32;
            FragBF fb;
#pragma unroll
            for (int v = 0; v < 8; ++v) {
                const int hh = v >> 2, j = v & 3;
                fb.u[v] = *(const unsigned*)(vr + 16 * hh + 8 * g + 2 * j);
            }
            oacc[t] = __builtin_amdgcn_wmma_f32_16x16x32_bf16(
                false, fa.v, false, fb.v, (short)0, oacc[t], false, false);
        }
    }
#pragma unroll
    for (int t = 0; t < 4; ++t) {
        const int n = h * DH + t * 16 + (lane & 15);
#pragma unroll
        for (int r = 0; r < 8; ++r) {
            const int row = rt * 16 + 8 * g + r;
            if (row < NTOK)
                o[((size_t)b * NTOK + row) * EE + n] = f32_to_bf16(oacc[t][r]);
        }
    }
}

// ---------------- host orchestration ----------------
extern "C" void kernel_launch(void* const* d_in, const int* in_sizes, int n_in,
                              void* d_out, int out_size, void* d_ws, size_t ws_size,
                              hipStream_t stream) {
    const float* img    = (const float*)d_in[0];
    const float* mu     = (const float*)d_in[1];
    const float* emb_w  = (const float*)d_in[2];
    const float* emb_b  = (const float*)d_in[3];
    const float* cls    = (const float*)d_in[4];
    const float* ln1_g  = (const float*)d_in[5];
    const float* ln1_b  = (const float*)d_in[6];
    const float* qkv_w  = (const float*)d_in[7];
    const float* proj_w = (const float*)d_in[8];
    const float* proj_b = (const float*)d_in[9];
    const float* ls1    = (const float*)d_in[10];
    const float* ln2_g  = (const float*)d_in[11];
    const float* ln2_b  = (const float*)d_in[12];
    const float* fc1_w  = (const float*)d_in[13];
    const float* fc1_b  = (const float*)d_in[14];
    const float* fc2_w  = (const float*)d_in[15];
    const float* fc2_b  = (const float*)d_in[16];
    const float* ls2    = (const float*)d_in[17];
    const float* lnf_g  = (const float*)d_in[18];
    const float* lnf_b  = (const float*)d_in[19];

    char* p = (char*)d_ws;
    auto carve = [&](size_t bytes) -> void* {
        void* r = (void*)p;
        p += (bytes + 255) & ~(size_t)255;
        return r;
    };
    unsigned short* emb_bf  = (unsigned short*)carve((size_t)IND * EE * 2);
    unsigned short* qkv_bf  = (unsigned short*)carve((size_t)12 * EE * QKVN * 2);
    unsigned short* proj_bf = (unsigned short*)carve((size_t)12 * EE * EE * 2);
    unsigned short* fc1_bf  = (unsigned short*)carve((size_t)12 * EE * HID3 * 2);
    unsigned short* fc2_bf  = (unsigned short*)carve((size_t)12 * HID3 * EE * 2);
    unsigned short* feats   = (unsigned short*)carve((size_t)BB * NF * IND * 2);
    float*          x       = (float*)carve((size_t)RROWS * EE * 4);
    unsigned short* h_bf    = (unsigned short*)carve((size_t)MPAD * EE * 2);
    unsigned short* qkv_act = (unsigned short*)carve((size_t)RROWS * QKVN * 2);
    unsigned short* vt_buf  = (unsigned short*)carve((size_t)BB * HEADS * DH * NPAD * 2);
    unsigned short* o_bf    = (unsigned short*)carve((size_t)MPAD * EE * 2);
    unsigned short* hid_bf  = (unsigned short*)carve((size_t)MPAD * HID3 * 2);

    auto pack = [&](const float* src, unsigned short* dst, int K, int N) {
        const int n = K * N;
        k_pack_B<<<(n + 255) / 256, 256, 0, stream>>>(src, dst, K, N);
    };
    pack(emb_w, emb_bf, IND, EE);
    for (int l = 0; l < 12; ++l) {
        pack(qkv_w  + (size_t)l * EE * QKVN, qkv_bf  + (size_t)l * EE * QKVN, EE, QKVN);
        pack(proj_w + (size_t)l * EE * EE,   proj_bf + (size_t)l * EE * EE,   EE, EE);
        pack(fc1_w  + (size_t)l * EE * HID3, fc1_bf  + (size_t)l * EE * HID3, EE, HID3);
        pack(fc2_w  + (size_t)l * HID3 * EE, fc2_bf  + (size_t)l * HID3 * EE, HID3, EE);
    }

    // token extraction + cls row
    k_extract_tokens<<<BB * NF, 128, 0, stream>>>(img, mu, feats);
    k_init_cls<<<(BB * EE + 255) / 256, 256, 0, stream>>>(cls, x);

    // embedding GEMM: [8192 x 2304] @ [2304 x 384] + bias -> x rows (cls remap)
    k_gemm_bf16<0><<<dim3(EE / 128, (BB * NF) / 128), 256, 0, stream>>>(
        feats, emb_bf, emb_b, nullptr, x, nullptr, nullptr,
        BB * NF, EE, IND, IND, EE, BB * NF);

    const int gy = MPAD / 128;   // 65
    for (int l = 0; l < 12; ++l) {
        const unsigned short* Wqkv = qkv_bf  + (size_t)l * EE * QKVN;
        const unsigned short* Wprj = proj_bf + (size_t)l * EE * EE;
        const unsigned short* Wf1  = fc1_bf  + (size_t)l * EE * HID3;
        const unsigned short* Wf2  = fc2_bf  + (size_t)l * HID3 * EE;

        k_layernorm<true><<<RROWS, 128, 0, stream>>>(x, ln1_g + l * EE, ln1_b + l * EE,
                                                     h_bf, nullptr);
        k_gemm_bf16<4><<<dim3(QKVN / 128, gy), 256, 0, stream>>>(
            h_bf, Wqkv, nullptr, nullptr, nullptr, qkv_act, vt_buf,
            RROWS, QKVN, EE, EE, QKVN, MPAD);
        k_attention<<<BB * HEADS * RT_PER_BH, 32, 0, stream>>>(qkv_act, vt_buf, o_bf);
        k_gemm_bf16<2><<<dim3(EE / 128, gy), 256, 0, stream>>>(
            o_bf, Wprj, proj_b + l * EE, ls1 + l * EE, x, nullptr, nullptr,
            RROWS, EE, EE, EE, EE, MPAD);
        k_layernorm<true><<<RROWS, 128, 0, stream>>>(x, ln2_g + l * EE, ln2_b + l * EE,
                                                     h_bf, nullptr);
        k_gemm_bf16<3><<<dim3(HID3 / 128, gy), 256, 0, stream>>>(
            h_bf, Wf1, fc1_b + l * HID3, nullptr, nullptr, hid_bf, nullptr,
            RROWS, HID3, EE, EE, HID3, MPAD);
        k_gemm_bf16<2><<<dim3(EE / 128, gy), 256, 0, stream>>>(
            hid_bf, Wf2, fc2_b + l * EE, ls2 + l * EE, x, nullptr, nullptr,
            RROWS, EE, HID3, HID3, EE, MPAD);
    }

    // final layernorm -> f32 output
    k_layernorm<false><<<RROWS, 128, 0, stream>>>(x, lnf_g, lnf_b,
                                                  nullptr, (float*)d_out);
}
